// Encoder_48979807043669
// MI455X (gfx1250) — matrix operand order repaired and checked by reading, not compiled, not run on hardware
//
#include <hip/hip_runtime.h>
#include <hip/hip_bf16.h>

// ---------------------------------------------------------------------------
// Sizes (fixed by the reference): N=2000, B=8, L=12, H=64, D=2, E=16000
//   NB = 16000 sequences, BL = 96 rows per node, Etot = 18000 edges (w/ loops)
// ---------------------------------------------------------------------------

typedef _Float16 h16 __attribute__((ext_vector_type(16)));
typedef _Float16 h8v __attribute__((ext_vector_type(8)));
typedef float    f8v __attribute__((ext_vector_type(8)));
typedef int      v4i __attribute__((vector_size(16)));

#if defined(__AMDGCN__) && __has_builtin(__builtin_amdgcn_global_load_async_to_lds_b128)
#define HAVE_ASYNC_LDS 1
#define AS1 __attribute__((address_space(1)))
#define AS3 __attribute__((address_space(3)))
#else
#define HAVE_ASYNC_LDS 0
#endif

__device__ __forceinline__ void wait_async_zero() {
#if defined(__AMDGCN__)
#if __has_builtin(__builtin_amdgcn_s_wait_asynccnt)
  __builtin_amdgcn_s_wait_asynccnt(0);
#else
  asm volatile("s_wait_asynccnt 0" ::: "memory");
#endif
#endif
}

__device__ __forceinline__ h16 cat8(h8v lo, h8v hi) {
  return __builtin_shufflevector(lo, hi, 0,1,2,3,4,5,6,7,8,9,10,11,12,13,14,15);
}

__device__ __forceinline__ f8v wmma_f16(h16 a, h16 b, f8v c) {
  // D = A(16x32 f16) * B(32x16 f16) + C(16x16 f32)
  return __builtin_amdgcn_wmma_f32_16x16x32_f16(
      /*neg_a=*/false, a, /*neg_b=*/false, b,
      /*c_mod=*/(short)0, c, /*reuse_a=*/false, /*reuse_b=*/false);
}

__device__ __forceinline__ float sigmoidf_(float x) {
  return 1.0f / (1.0f + __expf(-x));
}

// A-fragment (16x32, f16) from a [rows][64] f16 buffer laid out row-major.
// Layout: lanes0-15 hold M=lane, K=0..15; lanes16-31 K=16..31.
__device__ __forceinline__ h16 ldA(const _Float16* base, int row, int kt, int lane) {
  int kb = kt * 32 + ((lane < 16) ? 0 : 16);
  const _Float16* p = base + row * 64 + kb;
  h8v lo = *(const h8v*)p;
  h8v hi = *(const h8v*)(p + 8);
  return cat8(lo, hi);
}

// B-fragment (32x16, f16) from a [cols][64] f16 buffer (element [col][k]).
// Contiguous-k layout (same as ldA): lane holds column N=lane&15; lanes0-15 K=0..15, lanes16-31 K=16..31.
__device__ __forceinline__ h16 ldB(const _Float16* base, int col, int kt, int lane) {
  int k0 = kt * 32 + ((lane < 16) ? 0 : 16);
  const _Float16* p = base + col * 64 + k0;
  h8v lo = *(const h8v*)p;
  h8v hi = *(const h8v*)(p + 8);
  return cat8(lo, hi);
}

// ---------------------------------------------------------------------------
// Prep: transpose W3 columns / b3 into f16 [n][k] (n<64, k<128) so per-lane
// K-runs of the GAT B-fragments are contiguous.
// ---------------------------------------------------------------------------
__global__ void prep_w3_kernel(const float* __restrict__ W3, const float* __restrict__ b3,
                               _Float16* __restrict__ Ut, _Float16* __restrict__ Vt,
                               _Float16* __restrict__ B3t) {
  int i = blockIdx.x * 256 + threadIdx.x;   // i = n*128 + k
  if (i < 8192) {
    int n = i >> 7, k = i & 127;
    int src = k * 64 + n;                   // w flat index (k*H + n)
    Ut[i]  = (_Float16)W3[2 * src];
    Vt[i]  = (_Float16)W3[2 * src + 1];
    B3t[i] = (_Float16)b3[src];
  }
}

// ---------------------------------------------------------------------------
// GRU1: input dim D=2 (gi computed directly), gh via WMMA.
// Block = 128 threads = 4 waves, each wave owns 16 sequences.
// ---------------------------------------------------------------------------
__global__ __launch_bounds__(128) void gru1_kernel(
    const float* __restrict__ data,   // [16000][12][2]
    const float* __restrict__ Wih,    // [192][2]
    const float* __restrict__ Whh,    // [192][64]
    const float* __restrict__ bih,    // [192]
    const float* __restrict__ bhh,    // [192]
    _Float16* __restrict__ out1h,     // [16000][12][64] f16
    float* __restrict__ h1out)        // [16000][64] f32 (first half of d_out)
{
  __shared__ __align__(16) _Float16 wh_l[192 * 64];   // Whh as [col][k] f16
  __shared__ float wih_l[384];
  __shared__ float bih_l[192], bhh_l[192];
  __shared__ float xls[64 * 24];                      // 64 seqs x 12 x 2
  __shared__ __align__(16) _Float16 hbuf[4 * 16 * 64];

  int tid = threadIdx.x;
  for (int i = tid; i < 192 * 64; i += 128) wh_l[i] = (_Float16)Whh[i];
  for (int i = tid; i < 384; i += 128) wih_l[i] = Wih[i];
  for (int i = tid; i < 192; i += 128) { bih_l[i] = bih[i]; bhh_l[i] = bhh[i]; }
  {
    const float* dsrc = data + (size_t)blockIdx.x * 64 * 24;
    for (int i = tid; i < 64 * 24; i += 128) xls[i] = dsrc[i];
  }
  for (int i = tid; i < 4 * 16 * 64; i += 128) hbuf[i] = (_Float16)0.0f;
  __syncthreads();

  int wid = tid >> 5, lane = tid & 31, lane15 = lane & 15;
  int mhi = (lane < 16) ? 0 : 8;
  _Float16* hb = hbuf + wid * 1024;
  const float* xw = xls + wid * 16 * 24;
  int seqbase = blockIdx.x * 64 + wid * 16;

  for (int t = 0; t < 12; ++t) {
    h16 a[2];
#pragma unroll
    for (int kt = 0; kt < 2; ++kt) a[kt] = ldA(hb, lane15, kt, lane);

    f8v c[12];
#pragma unroll
    for (int nt = 0; nt < 12; ++nt) {
      int g = nt * 16 + lane15;
      if (nt < 8) {                       // r,z: fold gi + bih + bhh into C
        float w0 = wih_l[2 * g], w1 = wih_l[2 * g + 1];
        float bb = bih_l[g] + bhh_l[g];
#pragma unroll
        for (int v = 0; v < 8; ++v) {
          int m = mhi + v;
          c[nt][v] = xw[m * 24 + t * 2] * w0 + xw[m * 24 + t * 2 + 1] * w1 + bb;
        }
      } else {                            // n: C accumulates hn only (bhh)
        float bb = bhh_l[g];
#pragma unroll
        for (int v = 0; v < 8; ++v) c[nt][v] = bb;
      }
    }

#pragma unroll
    for (int nt = 0; nt < 12; ++nt)
#pragma unroll
      for (int kt = 0; kt < 2; ++kt)
        c[nt] = wmma_f16(a[kt], ldB(wh_l, nt * 16 + lane15, kt, lane), c[nt]);

    // gate nonlinearity + state update
#pragma unroll
    for (int j4 = 0; j4 < 4; ++j4) {
      int col = j4 * 16 + lane15;
      int gn = 128 + col;
      float wn0 = wih_l[2 * gn], wn1 = wih_l[2 * gn + 1], bn = bih_l[gn];
#pragma unroll
      for (int v = 0; v < 8; ++v) {
        int m = mhi + v;
        float r = sigmoidf_(c[j4][v]);
        float z = sigmoidf_(c[j4 + 4][v]);
        float inn = xw[m * 24 + t * 2] * wn0 + xw[m * 24 + t * 2 + 1] * wn1 + bn;
        float n = tanhf(inn + r * c[j4 + 8][v]);
        float hold = (float)hb[m * 64 + col];
        float hnew = (1.0f - z) * n + z * hold;
        hb[m * 64 + col] = (_Float16)hnew;
        out1h[(size_t)(seqbase + m) * 768 + t * 64 + col] = (_Float16)hnew;
        if (t == 11) h1out[(size_t)(seqbase + m) * 64 + col] = hnew;
      }
    }
    __syncthreads();
  }
}

// ---------------------------------------------------------------------------
// MetaGAT: one block (4 waves) per edge; wave w owns output N-tile w.
// Node state blocks (dst, src) are staged to LDS once per block via the
// gfx1250 async global->LDS path (ASYNCcnt), overlapped with the per-edge MLP;
// then A-fragments are ds_load_b128 from LDS (no 4x redundant global gathers).
// alpha = leaky_relu([S_i|S_j] @ (a*U + c*V + B3)); softmax over B; atomic agg.
// ---------------------------------------------------------------------------
__global__ __launch_bounds__(128) void gat_kernel(
    const int* __restrict__ ei,          // [2][16000]
    const float* __restrict__ features,  // [2000][64]
    const float* __restrict__ W1f, const float* __restrict__ b1,
    const float* __restrict__ W2f, const float* __restrict__ b2,
    const _Float16* __restrict__ out1h,  // [2000][96][64] f16
    const _Float16* __restrict__ Ut, const _Float16* __restrict__ Vt,
    const _Float16* __restrict__ B3t,
    float* __restrict__ agg)             // [2000][96][64] f32 (zeroed)
{
  __shared__ __align__(16) _Float16 sstage[2 * 6144];  // [0]=dst block, [1]=src block
  __shared__ float alphab[96 * 64];
  __shared__ float hdn1[16];
  __shared__ float ac[2];

  int e = blockIdx.x;
  int tid = threadIdx.x;
  int s, d;
  if (e < 16000) { s = ei[e]; d = ei[16000 + e]; }
  else           { s = e - 16000; d = s; }           // self loops

  // out1h is [seq][t][64] with seq = n*8+b -> node n's [96][64] block at n*6144
  const _Float16* pd = out1h + (size_t)d * 6144;
  const _Float16* ps = out1h + (size_t)s * 6144;

  // ---- stage both node blocks (2 x 12,288 B) into LDS ----
#if HAVE_ASYNC_LDS
  for (int i = tid; i < 1536; i += 128) {              // 16B chunks
    const _Float16* g = (i < 768 ? pd : ps) + (i & 767) * 8;
    _Float16* l = sstage + (i < 768 ? 0 : 6144) + (i & 767) * 8;
    __builtin_amdgcn_global_load_async_to_lds_b128(
        (AS1 v4i*)(__UINTPTR_TYPE__)g, (AS3 v4i*)(__UINTPTR_TYPE__)l, 0, 0);
  }
#else
  for (int i = tid; i < 1536; i += 128) {
    const h8v* g = (const h8v*)((i < 768 ? pd : ps) + (i & 767) * 8);
    *(h8v*)(sstage + (i < 768 ? 0 : 6144) + (i & 767) * 8) = *g;
  }
#endif

  // ---- per-edge weight-generating MLP (overlaps with the async DMA) ----
  if (tid < 16) {
    const float* fd = features + (size_t)d * 64;
    const float* fs = features + (size_t)s * 64;
    const float* w = W1f + tid * 128;
    float acc = b1[tid];
    for (int k = 0; k < 64; ++k) acc += fd[k] * w[k];
    for (int k = 0; k < 64; ++k) acc += fs[k] * w[64 + k];
    hdn1[tid] = sigmoidf_(acc);
  }
  __syncthreads();
  if (tid < 2) {
    float acc = b2[tid];
    for (int j = 0; j < 16; ++j) acc += hdn1[j] * W2f[tid * 16 + j];
    ac[tid] = sigmoidf_(acc);
  }
#if HAVE_ASYNC_LDS
  wait_async_zero();          // this wave's async loads complete
#endif
  __syncthreads();            // all waves' stage data + ac[] visible

  float av = ac[0], cv = ac[1];
  int wid = tid >> 5, lane = tid & 31, lane15 = lane & 15;
  int nt = wid;                                      // N-tile of this wave

  f8v c[6];
#pragma unroll
  for (int mt = 0; mt < 6; ++mt)
#pragma unroll
    for (int v = 0; v < 8; ++v) c[mt][v] = 0.0f;

#pragma unroll
  for (int kt = 0; kt < 4; ++kt) {
    // Synthesize B fragment: w[k][n] = a*U + c*V + B3  (16 contiguous k per lane)
    int n = nt * 16 + lane15;
    int k0 = kt * 32 + ((lane < 16) ? 0 : 16);
    const _Float16* up = Ut + n * 128 + k0;
    const _Float16* vp = Vt + n * 128 + k0;
    const _Float16* bp = B3t + n * 128 + k0;
    h16 u  = cat8(*(const h8v*)up, *(const h8v*)(up + 8));
    h16 vv = cat8(*(const h8v*)vp, *(const h8v*)(vp + 8));
    h16 bb = cat8(*(const h8v*)bp, *(const h8v*)(bp + 8));
    h16 bf;
#pragma unroll
    for (int i = 0; i < 16; ++i)
      bf[i] = (_Float16)(av * (float)u[i] + cv * (float)vv[i] + (float)bb[i]);

    // A fragments from LDS: K 0..63 = states_i (dst), 64..127 = states_j (src)
    const _Float16* node = sstage + ((kt < 2) ? 0 : 6144);
    int colb = (kt & 1) * 32 + ((lane < 16) ? 0 : 16);
#pragma unroll
    for (int mt = 0; mt < 6; ++mt) {
      const _Float16* p = node + (mt * 16 + lane15) * 64 + colb;
      h16 af = cat8(*(const h8v*)p, *(const h8v*)(p + 8));
      c[mt] = wmma_f16(af, bf, c[mt]);
    }
  }

  // leaky_relu -> LDS
  int mhi = (lane < 16) ? 0 : 8;
#pragma unroll
  for (int mt = 0; mt < 6; ++mt)
#pragma unroll
    for (int v = 0; v < 8; ++v) {
      float x = c[mt][v];
      float val = (x > 0.0f) ? x : 0.01f * x;
      int row = mt * 16 + mhi + v;
      alphab[row * 64 + nt * 16 + lane15] = val;
    }
  __syncthreads();

  // softmax over B (8 values, stride 12 rows) + message + atomic segment-sum
  for (int p = tid; p < 768; p += 128) {
    int l = p >> 6, h = p & 63;
    float vals[8];
    float mx = -1e30f;
#pragma unroll
    for (int b = 0; b < 8; ++b) {
      vals[b] = alphab[(b * 12 + l) * 64 + h];
      mx = fmaxf(mx, vals[b]);
    }
    float ssum = 0.0f;
#pragma unroll
    for (int b = 0; b < 8; ++b) { vals[b] = __expf(vals[b] - mx); ssum += vals[b]; }
    float inv = 1.0f / ssum;
#pragma unroll
    for (int b = 0; b < 8; ++b) {
      int row = b * 12 + l;
      float sj = (float)sstage[6144 + row * 64 + h];   // states_j from LDS
      float msg = vals[b] * inv * sj;
      __hip_atomic_fetch_add(&agg[(size_t)d * 6144 + row * 64 + h], msg,
                             __ATOMIC_RELAXED, __HIP_MEMORY_SCOPE_AGENT);
    }
  }
}

// ---------------------------------------------------------------------------
// relu(agg) -> f16 input for GRU2
// ---------------------------------------------------------------------------
__global__ void relu_cvt_kernel(const float* __restrict__ agg,
                                _Float16* __restrict__ outh, int n) {
  int i = blockIdx.x * 256 + threadIdx.x;
  if (i < n) {
    float x = agg[i];
    outh[i] = (_Float16)(x > 0.0f ? x : 0.0f);
  }
}

// ---------------------------------------------------------------------------
// GRU2: both gi (x@Wih2.T) and gh (h@Whh2.T) via WMMA.
// ---------------------------------------------------------------------------
__global__ __launch_bounds__(128) void gru2_kernel(
    const _Float16* __restrict__ xh,   // [16000][12][64] f16 (relu(agg))
    const float* __restrict__ Wih,     // [192][64]
    const float* __restrict__ Whh,     // [192][64]
    const float* __restrict__ bih, const float* __restrict__ bhh,
    float* __restrict__ h2out)         // [16000][64] f32 (second half of d_out)
{
  __shared__ __align__(16) _Float16 wi_l[192 * 64];
  __shared__ __align__(16) _Float16 wh_l[192 * 64];
  __shared__ float bih_l[192], bhh_l[192];
  __shared__ __align__(16) _Float16 hbuf[4 * 16 * 64];

  int tid = threadIdx.x;
  for (int i = tid; i < 192 * 64; i += 128) {
    wi_l[i] = (_Float16)Wih[i];
    wh_l[i] = (_Float16)Whh[i];
  }
  for (int i = tid; i < 192; i += 128) { bih_l[i] = bih[i]; bhh_l[i] = bhh[i]; }
  for (int i = tid; i < 4 * 16 * 64; i += 128) hbuf[i] = (_Float16)0.0f;
  __syncthreads();

  int wid = tid >> 5, lane = tid & 31, lane15 = lane & 15;
  int mhi = (lane < 16) ? 0 : 8;
  _Float16* hb = hbuf + wid * 1024;
  int seqbase = blockIdx.x * 64 + wid * 16;

  for (int t = 0; t < 12; ++t) {
    h16 ah[2], ax[2];
#pragma unroll
    for (int kt = 0; kt < 2; ++kt) {
      ah[kt] = ldA(hb, lane15, kt, lane);
      int kb = kt * 32 + ((lane < 16) ? 0 : 16);
      const _Float16* q = xh + (size_t)(seqbase + lane15) * 768 + t * 64 + kb;
      ax[kt] = cat8(*(const h8v*)q, *(const h8v*)(q + 8));
    }
    // prefetch next timestep's input row (speculative; dropped if OOB)
    __builtin_prefetch(xh + (size_t)(seqbase + lane15) * 768 + (t + 1) * 64, 0, 0);

    f8v crz[8], cgn[4], chn[4];
#pragma unroll
    for (int nt = 0; nt < 8; ++nt) {
      float bb = bih_l[nt * 16 + lane15] + bhh_l[nt * 16 + lane15];
#pragma unroll
      for (int v = 0; v < 8; ++v) crz[nt][v] = bb;
    }
#pragma unroll
    for (int j = 0; j < 4; ++j) {
      int g = 128 + j * 16 + lane15;
      float bi_ = bih_l[g], bh_ = bhh_l[g];
#pragma unroll
      for (int v = 0; v < 8; ++v) { cgn[j][v] = bi_; chn[j][v] = bh_; }
    }

#pragma unroll
    for (int nt = 0; nt < 8; ++nt)
#pragma unroll
      for (int kt = 0; kt < 2; ++kt) {
        crz[nt] = wmma_f16(ax[kt], ldB(wi_l, nt * 16 + lane15, kt, lane), crz[nt]);
        crz[nt] = wmma_f16(ah[kt], ldB(wh_l, nt * 16 + lane15, kt, lane), crz[nt]);
      }
#pragma unroll
    for (int j = 0; j < 4; ++j)
#pragma unroll
      for (int kt = 0; kt < 2; ++kt) {
        int col = (8 + j) * 16 + lane15;
        cgn[j] = wmma_f16(ax[kt], ldB(wi_l, col, kt, lane), cgn[j]);
        chn[j] = wmma_f16(ah[kt], ldB(wh_l, col, kt, lane), chn[j]);
      }

#pragma unroll
    for (int j4 = 0; j4 < 4; ++j4) {
      int col = j4 * 16 + lane15;
#pragma unroll
      for (int v = 0; v < 8; ++v) {
        int m = mhi + v;
        float r = sigmoidf_(crz[j4][v]);
        float z = sigmoidf_(crz[j4 + 4][v]);
        float n = tanhf(cgn[j4][v] + r * chn[j4][v]);
        float hold = (float)hb[m * 64 + col];
        float hnew = (1.0f - z) * n + z * hold;
        hb[m * 64 + col] = (_Float16)hnew;
        if (t == 11) h2out[(size_t)(seqbase + m) * 64 + col] = hnew;
      }
    }
    __syncthreads();
  }
}

// ---------------------------------------------------------------------------
// Launch
// ---------------------------------------------------------------------------
extern "C" void kernel_launch(void* const* d_in, const int* in_sizes, int n_in,
                              void* d_out, int out_size, void* d_ws, size_t ws_size,
                              hipStream_t stream) {
  const float* data     = (const float*)d_in[0];
  const float* features = (const float*)d_in[1];
  const int*   ei       = (const int*)d_in[2];
  const float* Wih1 = (const float*)d_in[3];
  const float* Whh1 = (const float*)d_in[4];
  const float* bih1 = (const float*)d_in[5];
  const float* bhh1 = (const float*)d_in[6];
  const float* W1   = (const float*)d_in[7];
  const float* b1   = (const float*)d_in[8];
  const float* W2   = (const float*)d_in[9];
  const float* b2   = (const float*)d_in[10];
  const float* W3   = (const float*)d_in[11];
  const float* b3   = (const float*)d_in[12];
  const float* Wih2 = (const float*)d_in[13];
  const float* Whh2 = (const float*)d_in[14];
  const float* bih2 = (const float*)d_in[15];
  const float* bhh2 = (const float*)d_in[16];
  float* out = (float*)d_out;

  // Workspace layout (all offsets 16B aligned), total ~98.4 MB:
  char* ws = (char*)d_ws;
  _Float16* out1h = (_Float16*)(ws);                    // 24,576,000 B
  _Float16* out2h = (_Float16*)(ws + 24576000);         // 24,576,000 B
  float*    agg   = (float*)(ws + 49152000);            // 49,152,000 B
  _Float16* Ut    = (_Float16*)(ws + 98304000);         // 16,384 B
  _Float16* Vt    = (_Float16*)(ws + 98304000 + 16384); // 16,384 B
  _Float16* B3t   = (_Float16*)(ws + 98304000 + 32768); // 16,384 B

  prep_w3_kernel<<<32, 256, 0, stream>>>(W3, b3, Ut, Vt, B3t);
  (void)hipMemsetAsync(agg, 0, 49152000, stream);
  gru1_kernel<<<250, 128, 0, stream>>>(data, Wih1, Whh1, bih1, bhh1, out1h, out);
  gat_kernel<<<18000, 128, 0, stream>>>(ei, features, W1, b1, W2, b2,
                                        out1h, Ut, Vt, B3t, agg);
  relu_cvt_kernel<<<48000, 256, 0, stream>>>(agg, out2h, 12288000);
  gru2_kernel<<<250, 128, 0, stream>>>(out2h, Wih2, Whh2, bih2, bhh2,
                                       out + 1024000);
}
